// DeepSeekHybridAttention_70772471103968
// MI455X (gfx1250) — matrix operand order repaired and checked by reading, not compiled, not run on hardware
//
#include <hip/hip_runtime.h>
#include <hip/hip_bf16.h>
#include <math.h>
#include <stdint.h>

// ---------------- problem constants ----------------
constexpr int cB     = 4;
constexpr int cT     = 2048;
constexpr int cC     = 1024;
constexpr int cH     = 8;
constexpr int cHKV   = 4;
constexpr int cD     = 128;
constexpr int cRATIO = 4;
constexpr int cG     = cT / cRATIO;   // 512
constexpr int cHI    = 8;
constexpr int cDI    = 64;
constexpr int cTOPK  = 64;
constexpr int cRANK  = 512;
constexpr float cNEG   = -1e30f;
constexpr float cTHETA = 160000.0f;

// ---------------- types ----------------
typedef __attribute__((ext_vector_type(16))) __bf16 v16bf;
typedef __attribute__((ext_vector_type(8)))  __bf16 v8bf;
typedef __attribute__((ext_vector_type(8)))  float  v8f;
typedef __attribute__((ext_vector_type(4)))  unsigned int v4u;
typedef __attribute__((ext_vector_type(8)))  int v8i;
typedef __attribute__((ext_vector_type(4)))  int v4i;

union FragU { v16bf v; v8bf h[2]; };

__device__ __forceinline__ unsigned short f2bf(float f) {
  unsigned int u = __float_as_uint(f);
  u += 0x7fffu + ((u >> 16) & 1u);     // round-to-nearest-even
  return (unsigned short)(u >> 16);
}

__device__ __forceinline__ v8f wmma_bf16(v16bf a, v16bf b, v8f c) {
  // D = A(16x32 bf16) * B(32x16 bf16) + C(16x16 f32)
  return __builtin_amdgcn_wmma_f32_16x16x32_bf16(false, a, false, b, (short)0, c,
                                                 false, false);
}

// A-fragment (16x32, M in lanes): lane<16 holds K0..15, lane>=16 K16..31
__device__ __forceinline__ v16bf ldfragA(const unsigned short* rowp, int akb) {
  FragU u;
  u.h[0] = *(const v8bf*)(rowp + akb);
  u.h[1] = *(const v8bf*)(rowp + akb + 8);
  return u.v;
}
// B-fragment (32x16, N in lanes): staged n-major in LDS; lane<16 K0..15, lane>=16 K16..31
__device__ __forceinline__ v16bf ldfragB(const unsigned short* rowp, int bkb) {
  FragU u;
  u.h[0] = *(const v8bf*)(rowp + bkb);
  u.h[1] = *(const v8bf*)(rowp + bkb + 8);
  return u.v;
}

// ---------------- Tensor Data Mover: 2D f32 tile global -> LDS ----------------
// D# per CDNA5 ISA: group0 {count=1, lds_addr, global_addr,
// type=2}, group1 {data_size=4B, tensor_dim0/1, tile_dim0/1, dim0 row stride}.
// 6-arg builtin form (clang-23 / therock headers): (v4u, v8i, v4i, v4i, v8i, i32).
__device__ __forceinline__ void tdm_load_tile_f32(
    unsigned int lds_byte_off, const float* gptr,
    unsigned int tile_w, unsigned int tile_h,
    unsigned int tensor_w, unsigned int tensor_h,
    unsigned long long row_stride_elems) {
  unsigned long long ga = (unsigned long long)(uintptr_t)gptr;
  v4u g0;
  g0[0] = 1u;                                              // count=1, user mode
  g0[1] = lds_byte_off;                                    // lds_addr
  g0[2] = (unsigned int)ga;                                // global_addr[31:0]
  g0[3] = (unsigned int)((ga >> 32) & 0x01ffffffu) | (2u << 30);  // addr[56:32]|type=2
  v8i g1;
  g1[0] = (int)(2u << 16);                                 // data_size = 4 bytes
  g1[1] = (int)((tensor_w & 0xffffu) << 16);               // tensor_dim0[15:0]
  g1[2] = (int)(((tensor_w >> 16) & 0xffffu) | ((tensor_h & 0xffffu) << 16));
  g1[3] = (int)(((tensor_h >> 16) & 0xffffu) | (tile_w << 16));   // tile_dim0
  g1[4] = (int)tile_h;                                     // tile_dim1 (tile_dim2=0)
  g1[5] = (int)(unsigned int)(row_stride_elems & 0xffffffffu);    // dim0_stride lo
  g1[6] = (int)(unsigned int)((row_stride_elems >> 32) & 0xffffu);// dim0_stride hi
  g1[7] = 0;
  v4i z4 = {0, 0, 0, 0};
  v8i z8 = {0, 0, 0, 0, 0, 0, 0, 0};
  __builtin_amdgcn_tensor_load_to_lds(g0, g1, z4, z4, z8, 0);
}

// ---------------- generic bf16 WMMA GEMM ----------------
// C[M,N] = alpha * A[M,K] @ (transB ? B[N,K]^T : B[K,N])
// Block tile 64x128, 8 waves, 2x2 16x16 tiles per wave (4 WMMA / k-step).
// A tile streamed by the Tensor Data Mover, converted f32->bf16 in LDS.
__global__ __launch_bounds__(256) void gemm_bf16_wmma(
    const float* __restrict__ A, const float* __restrict__ Bg,
    float* __restrict__ C, int M, int N, int K, int transB, float alpha) {
  __shared__ float          As32[64 * 32];   // TDM staging (f32)
  __shared__ unsigned short As[64][40];      // [m][k] bf16
  __shared__ unsigned short Bs[128][40];     // [n][k] bf16
  const int tid  = threadIdx.x;
  const int lane = tid & 31;
  const int wave = tid >> 5;
  const int bm = blockIdx.x * 64;
  const int bn = blockIdx.y * 128;
  const int mt0 = (wave & 1) * 2;            // m-tiles {mt0, mt0+1}
  const int nt0 = (wave >> 1) * 2;           // n-tiles {nt0, nt0+1}
  const int mrow = lane & 15;
  const int halfsel = lane >> 4;
  const int akb = halfsel * 16;
  const int bkb = halfsel * 16;
  v8f acc00 = {}, acc01 = {}, acc10 = {}, acc11 = {};

  for (int k0 = 0; k0 < K; k0 += 32) {
    // async A tile via TDM: 32 elems/row x 64 rows, row stride K
    if (tid == 0)
      tdm_load_tile_f32((unsigned int)(uintptr_t)As32,
                        A + (size_t)bm * K + k0, 32u, 64u,
                        (unsigned int)K, (unsigned int)M,
                        (unsigned long long)K);
    if (k0 + 32 < K)
      __builtin_prefetch(A + (size_t)(bm + (tid >> 2)) * K + k0 + 32, 0, 1);
    // B tile -> Bs[n][k] (bf16)
    if (transB) {
      #pragma unroll
      for (int i = 0; i < 4; ++i) {
        int lin = tid + i * 256;
        int n = lin >> 3;
        int c = (lin & 7) * 4;
        const float4 f = *(const float4*)(Bg + (size_t)(bn + n) * K + k0 + c);
        Bs[n][c + 0] = f2bf(f.x); Bs[n][c + 1] = f2bf(f.y);
        Bs[n][c + 2] = f2bf(f.z); Bs[n][c + 3] = f2bf(f.w);
      }
    } else {
      #pragma unroll
      for (int i = 0; i < 4; ++i) {
        int lin = tid + i * 256;
        int kk = lin >> 5;
        int c = (lin & 31) * 4;
        const float4 f = *(const float4*)(Bg + (size_t)(k0 + kk) * N + bn + c);
        Bs[c + 0][kk] = f2bf(f.x); Bs[c + 1][kk] = f2bf(f.y);
        Bs[c + 2][kk] = f2bf(f.z); Bs[c + 3][kk] = f2bf(f.w);
      }
    }
    if (tid == 0) __builtin_amdgcn_s_wait_tensorcnt(0);
    __syncthreads();                                  // As32 + Bs visible
    // convert staged A tile to bf16
    #pragma unroll
    for (int i = 0; i < 8; ++i) {
      int lin = tid + i * 256;
      As[lin >> 5][lin & 31] = f2bf(As32[lin]);
    }
    __syncthreads();                                  // As ready
    v16bf a0 = ldfragA(&As[(mt0 + 0) * 16 + mrow][0], akb);
    v16bf a1 = ldfragA(&As[(mt0 + 1) * 16 + mrow][0], akb);
    v16bf b0 = ldfragB(&Bs[(nt0 + 0) * 16 + mrow][0], bkb);
    v16bf b1 = ldfragB(&Bs[(nt0 + 1) * 16 + mrow][0], bkb);
    acc00 = wmma_bf16(a0, b0, acc00);
    acc01 = wmma_bf16(a0, b1, acc01);
    acc10 = wmma_bf16(a1, b0, acc10);
    acc11 = wmma_bf16(a1, b1, acc11);
    __syncthreads();                                  // before overwrite
  }
  const int col0 = bn + (nt0 + 0) * 16 + mrow;
  const int col1 = bn + (nt0 + 1) * 16 + mrow;
  const int r0 = bm + (mt0 + 0) * 16 + halfsel * 8;
  const int r1 = bm + (mt0 + 1) * 16 + halfsel * 8;
  #pragma unroll
  for (int v = 0; v < 8; ++v) {
    C[(size_t)(r0 + v) * N + col0] = acc00[v] * alpha;
    C[(size_t)(r0 + v) * N + col1] = acc01[v] * alpha;
    C[(size_t)(r1 + v) * N + col0] = acc10[v] * alpha;
    C[(size_t)(r1 + v) * N + col1] = acc11[v] * alpha;
  }
}

// ---------------- gated compress (ck/cv) ----------------
__global__ __launch_bounds__(256) void compress_ckcv(
    const float* __restrict__ xk, const float* __restrict__ xv,
    const float* __restrict__ xg, const float* __restrict__ ape,
    float* __restrict__ ck, float* __restrict__ cv) {
  int idx = blockIdx.x * 256 + threadIdx.x;      // over B*G*HKV*D
  int d = idx % cD;
  int h = (idx / cD) % cHKV;
  int g = (idx / (cD * cHKV)) % cG;
  int b = idx / (cD * cHKV * cG);
  float kv[4], vv[4], gv[4];
  float mx = cNEG;
  #pragma unroll
  for (int r = 0; r < cRATIO; ++r) {
    size_t off = (size_t)(b * cT + g * cRATIO + r) * (cHKV * cD) + h * cD + d;
    kv[r] = xk[off]; vv[r] = xv[off];
    gv[r] = xg[off] + ape[(r * cHKV + h) * cD + d];
    mx = fmaxf(mx, gv[r]);
  }
  float s = 0.f;
  #pragma unroll
  for (int r = 0; r < cRATIO; ++r) { gv[r] = __expf(gv[r] - mx); s += gv[r]; }
  float inv = 1.f / s, ok = 0.f, ov = 0.f;
  #pragma unroll
  for (int r = 0; r < cRATIO; ++r) { float w = gv[r] * inv; ok += kv[r] * w; ov += vv[r] * w; }
  size_t o = (size_t)((b * cG + g) * cHKV + h) * cD + d;
  ck[o] = ok; cv[o] = ov;
}

// ---------------- gated compress + rmsnorm (index keys) ----------------
__global__ __launch_bounds__(64) void compress_ikeys(
    const float* __restrict__ xk, const float* __restrict__ xg,
    const float* __restrict__ ape, float* __restrict__ ikeys) {
  __shared__ float red[64];
  int bgh = blockIdx.x;                // over B*G*HI
  int d = threadIdx.x;
  int h = bgh % cHI;
  int g = (bgh / cHI) % cG;
  int b = bgh / (cHI * cG);
  float kv[4], gv[4];
  float mx = cNEG;
  #pragma unroll
  for (int r = 0; r < cRATIO; ++r) {
    size_t off = (size_t)(b * cT + g * cRATIO + r) * (cHI * cDI) + h * cDI + d;
    kv[r] = xk[off];
    gv[r] = xg[off] + ape[(r * cHI + h) * cDI + d];
    mx = fmaxf(mx, gv[r]);
  }
  float s = 0.f;
  #pragma unroll
  for (int r = 0; r < cRATIO; ++r) { gv[r] = __expf(gv[r] - mx); s += gv[r]; }
  float inv = 1.f / s, o = 0.f;
  #pragma unroll
  for (int r = 0; r < cRATIO; ++r) o += kv[r] * (gv[r] * inv);
  red[d] = o * o;
  __syncthreads();
  for (int st = 32; st > 0; st >>= 1) { if (d < st) red[d] += red[d + st]; __syncthreads(); }
  float scale = rsqrtf(red[0] / (float)cDI + 1e-6f);
  ikeys[(size_t)((b * cG + g) * cHI + h) * cDI + d] = o * scale;
}

// ---------------- generic in-place row rmsnorm ----------------
__global__ void rmsnorm_rows(float* __restrict__ X, int cols) {
  __shared__ float red[256];
  const int row = blockIdx.x, tid = threadIdx.x, nt = blockDim.x;
  float ss = 0.f;
  for (int c = tid; c < cols; c += nt) { float v = X[(size_t)row * cols + c]; ss += v * v; }
  red[tid] = ss;
  __syncthreads();
  for (int s = nt >> 1; s > 0; s >>= 1) { if (tid < s) red[tid] += red[tid + s]; __syncthreads(); }
  float scale = rsqrtf(red[0] / (float)cols + 1e-6f);
  for (int c = tid; c < cols; c += nt) X[(size_t)row * cols + c] *= scale;
}

// ---------------- partial RoPE (in place, last 64 of 128 dims) ----------------
__global__ __launch_bounds__(256) void rope_inplace(float* __restrict__ X, int Tdim,
                                                    int nh, int pos_mult, int pos_add) {
  int idx = blockIdx.x * 256 + threadIdx.x;   // over B*Tdim*nh*32
  int d0 = idx & 31;
  int h  = (idx >> 5) % nh;
  int t  = (idx / (32 * nh)) % Tdim;
  int b  = idx / (32 * nh * Tdim);
  float pos = (float)(t * pos_mult + pos_add);
  float invf = __expf(-(float)d0 * (logf(cTHETA) / 32.f));
  float ang = pos * invf;
  float c = __cosf(ang), s = __sinf(ang);
  size_t base = ((size_t)(b * Tdim + t) * nh + h) * cD;
  float x1 = X[base + 64 + d0], x2 = X[base + 96 + d0];
  X[base + 64 + d0] = x1 * c - x2 * s;
  X[base + 96 + d0] = x1 * s + x2 * c;
}

// ---------------- causal + stable top-k selection mask ----------------
__global__ __launch_bounds__(256) void topk_mask_kernel(
    const float* __restrict__ isc, unsigned char* __restrict__ mask) {
  __shared__ float vals[cG];
  const int bt = blockIdx.x;            // b*T + t
  const int t = bt % cT;
  const float* row = isc + (size_t)bt * cG;
  for (int g = threadIdx.x; g < cG; g += 256)
    vals[g] = (g * cRATIO + (cRATIO - 1) <= t) ? row[g] : cNEG;
  __syncthreads();
  for (int g = threadIdx.x; g < cG; g += 256) {
    float v = vals[g];
    int rank = 0;
    for (int j = 0; j < cG; ++j) {
      float u = vals[j];
      rank += (u > v) || (u == v && j < g);       // stable tie-break == lax.top_k
    }
    bool causal = (g * cRATIO + (cRATIO - 1)) <= t;
    mask[(size_t)bt * cG + g] = (rank < cTOPK && causal) ? 1 : 0;
  }
}

// ---------------- fused masked attention: S=QK^T, softmax, O=PV ----------------
__global__ __launch_bounds__(256) void attn_fused_kernel(
    const float* __restrict__ q, const float* __restrict__ ck,
    const float* __restrict__ cv, const unsigned char* __restrict__ mask,
    float* __restrict__ attn) {
  constexpr int QT = 32;
  constexpr int QLD = 136, KLD = 136, VLD = 72, SLD = 512, PLD = 520;
  extern __shared__ unsigned char smem[];
  unsigned short* Qs  = (unsigned short*)smem;           // [QT][QLD] bf16
  unsigned short* Ks  = Qs + QT * QLD;                   // [64][KLD] bf16, [g][d]
  unsigned short* Vs  = Ks + 64 * KLD;                   // [128][VLD] bf16, [d][g]
  float*          Ss  = (float*)(Vs + 128 * VLD);        // [QT][SLD] f32 scores
  unsigned short* Ps  = (unsigned short*)(Ss + QT * SLD);// [QT][PLD] bf16 probs
  float*          red = (float*)(Ps + QT * PLD);         // [QT][8]
  float*          KV32 = red + QT * 8;                   // [64][128] f32 TDM staging

  const int tb = blockIdx.x, h = blockIdx.y, b = blockIdx.z;
  const int hk = h >> 1;                                 // rep = H/HKV = 2
  const int t0 = tb * QT;
  const int tid = threadIdx.x;
  const int lane = tid & 31;
  const int wave = tid >> 5;
  const int mrow = lane & 15;
  const int halfsel = lane >> 4;
  const int akb = halfsel * 16;
  const int bkb = halfsel * 16;
  const float scale = 0.088388347648318447f;             // D^-0.5

  // stage Q tile (QT x 128) as bf16
  #pragma unroll
  for (int i = 0; i < 4; ++i) {
    int lin = tid + i * 256;
    int r = lin >> 5;
    int c = (lin & 31) * 4;
    const float4 f = *(const float4*)(q + ((size_t)(b * cT + t0 + r) * cH + h) * cD + c);
    Qs[r * QLD + c + 0] = f2bf(f.x); Qs[r * QLD + c + 1] = f2bf(f.y);
    Qs[r * QLD + c + 2] = f2bf(f.z); Qs[r * QLD + c + 3] = f2bf(f.w);
  }
  __syncthreads();

  // Phase 1: scores S[QT][G], K chunks streamed via TDM
  {
    const int mt = wave >> 2, nt = wave & 3;
    for (int gc = 0; gc < cG; gc += 64) {
      if (tid == 0) {
        tdm_load_tile_f32((unsigned int)(uintptr_t)KV32,
                          ck + ((size_t)(b * cG + gc) * cHKV + hk) * cD,
                          128u, 64u, 128u, (unsigned int)cG,
                          (unsigned long long)(cHKV * cD));
        __builtin_amdgcn_s_wait_tensorcnt(0);
      }
      __syncthreads();
      #pragma unroll
      for (int i = 0; i < 32; ++i) {          // 64x128 f32 -> bf16 [g][d]
        int lin = tid + i * 256;
        Ks[(lin >> 7) * KLD + (lin & 127)] = f2bf(KV32[lin]);
      }
      __syncthreads();
      v8f s = {};
      #pragma unroll
      for (int kk = 0; kk < cD; kk += 32) {
        v16bf af = ldfragA(&Qs[(mt * 16 + mrow) * QLD + kk], akb);
        v16bf bf = ldfragB(&Ks[(nt * 16 + mrow) * KLD + kk], bkb);
        s = wmma_bf16(af, bf, s);
      }
      const int col = gc + nt * 16 + mrow;
      const int rb = mt * 16 + halfsel * 8;
      #pragma unroll
      for (int v = 0; v < 8; ++v) Ss[(rb + v) * SLD + col] = s[v] * scale;
      __syncthreads();
    }
  }

  // Phase 2: masked softmax; stage P as bf16 in A-fragment-friendly layout
  {
    const int r = tid >> 3, sub = tid & 7;
    const unsigned char* mp = mask + (size_t)(b * cT + t0 + r) * cG;
    float mx = cNEG;
    for (int j = 0; j < 64; ++j) {
      int g0 = sub * 64 + j;
      float sv = mp[g0] ? Ss[r * SLD + g0] : cNEG;
      mx = fmaxf(mx, sv);
    }
    red[r * 8 + sub] = mx;
    __syncthreads();
    float rowmax = cNEG;
    #pragma unroll
    for (int j = 0; j < 8; ++j) rowmax = fmaxf(rowmax, red[r * 8 + j]);
    __syncthreads();
    float sm = 0.f;
    for (int j = 0; j < 64; ++j) {
      int g0 = sub * 64 + j;
      float sv = mp[g0] ? Ss[r * SLD + g0] : cNEG;
      sm += __expf(sv - rowmax);
    }
    red[r * 8 + sub] = sm;
    __syncthreads();
    float rowsum = 0.f;
    #pragma unroll
    for (int j = 0; j < 8; ++j) rowsum += red[r * 8 + j];
    float inv = 1.f / rowsum;
    for (int j = 0; j < 64; ++j) {
      int g0 = sub * 64 + j;
      float p = mp[g0] ? __expf(Ss[r * SLD + g0] - rowmax) * inv : 0.f;
      Ps[r * PLD + g0] = f2bf(p);
    }
  }
  __syncthreads();

  // Phase 3: O = P @ V, V chunks streamed via TDM then transposed to [d][g]
  {
    const int mt = wave >> 2;
    const int nt0 = (wave & 3) * 2;
    v8f o0 = {}, o1 = {};
    for (int gc = 0; gc < cG; gc += 64) {
      if (tid == 0) {
        tdm_load_tile_f32((unsigned int)(uintptr_t)KV32,
                          cv + ((size_t)(b * cG + gc) * cHKV + hk) * cD,
                          128u, 64u, 128u, (unsigned int)cG,
                          (unsigned long long)(cHKV * cD));
        __builtin_amdgcn_s_wait_tensorcnt(0);
      }
      __syncthreads();
      #pragma unroll
      for (int i = 0; i < 32; ++i) {          // [g][d] f32 -> [d][g] bf16
        int lin = tid + i * 256;
        Vs[(lin & 127) * VLD + (lin >> 7)] = f2bf(KV32[lin]);
      }
      __syncthreads();
      #pragma unroll
      for (int kk = 0; kk < 64; kk += 32) {
        v16bf af = ldfragA(&Ps[(mt * 16 + mrow) * PLD + gc + kk], akb);
        v16bf b0 = ldfragB(&Vs[(nt0 * 16 + mrow) * VLD + kk], bkb);
        v16bf b1 = ldfragB(&Vs[((nt0 + 1) * 16 + mrow) * VLD + kk], bkb);
        o0 = wmma_bf16(af, b0, o0);
        o1 = wmma_bf16(af, b1, o1);
      }
      __syncthreads();
    }
    const int rb = t0 + mt * 16 + halfsel * 8;
    #pragma unroll
    for (int v = 0; v < 8; ++v) {
      size_t rowoff = (size_t)(b * cT + rb + v) * cC + h * cD;
      attn[rowoff + nt0 * 16 + mrow]       = o0[v];
      attn[rowoff + (nt0 + 1) * 16 + mrow] = o1[v];
    }
  }
}

// ---------------- host launcher ----------------
static void launch_gemm(const float* A, const float* Bm, float* Cm, int M, int N,
                        int K, int transB, float alpha, hipStream_t s) {
  dim3 grid(M / 64, N / 128);
  gemm_bf16_wmma<<<grid, 256, 0, s>>>(A, Bm, Cm, M, N, K, transB, alpha);
}

extern "C" void kernel_launch(void* const* d_in, const int* in_sizes, int n_in,
                              void* d_out, int out_size, void* d_ws, size_t ws_size,
                              hipStream_t stream) {
  (void)in_sizes; (void)n_in; (void)out_size; (void)ws_size;
  const float* x     = (const float*)d_in[0];
  const float* Wq    = (const float*)d_in[1];
  const float* Wck   = (const float*)d_in[2];
  const float* Wcv   = (const float*)d_in[3];
  const float* Wcg   = (const float*)d_in[4];
  const float* ape_c = (const float*)d_in[5];
  const float* Wiq   = (const float*)d_in[6];
  const float* Wik   = (const float*)d_in[7];
  const float* Wig   = (const float*)d_in[8];
  const float* ape_i = (const float*)d_in[9];
  const float* Wa    = (const float*)d_in[10];
  const float* Wb    = (const float*)d_in[11];
  float* out = (float*)d_out;

  const size_t MT = (size_t)cB * cT;        // 8192 rows
  float* ws = (float*)d_ws;
  float* xq    = ws;  ws += MT * cC;                      // 8M f
  float* xck   = ws;  ws += MT * (cHKV * cD);             // 4M f
  float* xcv   = ws;  ws += MT * (cHKV * cD);
  float* xcg   = ws;  ws += MT * (cHKV * cD);
  float* xik   = ws;  ws += MT * (cHI * cDI);
  float* xig   = ws;  ws += MT * (cHI * cDI);
  float* xiq   = ws;  ws += MT * (cHI * cDI);
  float* ckb   = ws;  ws += (size_t)cB * cG * cHKV * cD;  // 1M f
  float* cvb   = ws;  ws += (size_t)cB * cG * cHKV * cD;
  float* ikeys = ws;  ws += (size_t)cB * cG * cHI * cDI;  // 1M f
  float* isc   = ws;  ws += (size_t)cB * cT * cG;         // 4M f
  unsigned char* mask = (unsigned char*)ws;
  ws += ((size_t)cB * cT * cG) / sizeof(float);           // 4MB
  float* attn = xck;   // alias: 8M floats over xck+xcv (both dead by then)
  float* rbuf = xig;   // alias: 4M floats (dead by then)

  // 1) projections (WMMA GEMMs, TDM-fed A tiles)
  launch_gemm(x, Wq,  xq,  (int)MT, cC,          cC, 0, 1.f, stream);
  launch_gemm(x, Wck, xck, (int)MT, cHKV * cD,   cC, 0, 1.f, stream);
  launch_gemm(x, Wcv, xcv, (int)MT, cHKV * cD,   cC, 0, 1.f, stream);
  launch_gemm(x, Wcg, xcg, (int)MT, cHKV * cD,   cC, 0, 1.f, stream);
  launch_gemm(x, Wik, xik, (int)MT, cHI * cDI,   cC, 0, 1.f, stream);
  launch_gemm(x, Wig, xig, (int)MT, cHI * cDI,   cC, 0, 1.f, stream);
  launch_gemm(x, Wiq, xiq, (int)MT, cHI * cDI,   cC, 0, 1.f, stream);

  // 2) gated compression + norms
  compress_ckcv<<<(cB * cG * cHKV * cD) / 256, 256, 0, stream>>>(xck, xcv, xcg,
                                                                 ape_c, ckb, cvb);
  compress_ikeys<<<cB * cG * cHI, 64, 0, stream>>>(xik, xig, ape_i, ikeys);
  rmsnorm_rows<<<cB * cT * cHI, 64, 0, stream>>>(xiq, cDI);

  // 3) RoPE on q and compressed keys
  rope_inplace<<<(cB * cT * cH * 32) / 256, 256, 0, stream>>>(xq, cT, cH, 1, 0);
  rope_inplace<<<(cB * cG * cHKV * 32) / 256, 256, 0, stream>>>(ckb, cG, cHKV,
                                                                cRATIO, cRATIO - 1);

  // 4) index scores: mean-over-heads == (1/HI) * full-512 dot; alpha = DI^-0.5 / HI
  for (int b = 0; b < cB; ++b) {
    launch_gemm(xiq + (size_t)b * cT * (cHI * cDI),
                ikeys + (size_t)b * cG * (cHI * cDI),
                isc + (size_t)b * cT * cG,
                cT, cG, cHI * cDI, 1, 1.0f / 64.0f, stream);
  }

  // 5) causal + top-k selection mask
  topk_mask_kernel<<<cB * cT, 256, 0, stream>>>(isc, mask);

  // 6) fused masked attention (TDM K/V streaming, WMMA QK^T + softmax + WMMA PV)
  {
    size_t smem = (size_t)(32 * 136 + 64 * 136 + 128 * 72) * 2  // Qs,Ks,Vs bf16
                + (size_t)32 * 512 * 4                          // Ss f32
                + (size_t)32 * 520 * 2                          // Ps bf16
                + (size_t)32 * 8 * 4                            // red
                + (size_t)64 * 128 * 4;                         // KV32 staging
    attn_fused_kernel<<<dim3(cT / 32, cH, cB), 256, smem, stream>>>(xq, ckb, cvb,
                                                                    mask, attn);
  }

  // 7) low-rank output: r = attn @ Wa^T, rmsnorm, out = rn @ Wb
  launch_gemm(attn, Wa, rbuf, (int)MT, cRANK, cC, 1, 1.f, stream);
  rmsnorm_rows<<<(int)MT, 256, 0, stream>>>(rbuf, cRANK);
  launch_gemm(rbuf, Wb, out, (int)MT, cC, cRANK, 0, 1.f, stream);
}
